// BatchAttentionLayer_82437602279874
// MI455X (gfx1250) — matrix operand order repaired and checked by reading, not compiled, not run on hardware
//
#include <hip/hip_runtime.h>
#include <hip/hip_bf16.h>

typedef __attribute__((ext_vector_type(16))) _Float16 v16h;
typedef __attribute__((ext_vector_type(8)))  _Float16 v8h;
typedef __attribute__((ext_vector_type(8)))  float    v8f;

#define S_DIM  512
#define T_DIM  512
#define DU     512
#define D_DIM  1024

// ---- compile-time probes (report in stderr, no codegen impact) -------------
#if defined(__HIP_DEVICE_COMPILE__)
#if __has_builtin(__builtin_amdgcn_tanhf)
#warning CDNA5_PROBE: native __builtin_amdgcn_tanhf AVAILABLE
#else
#warning CDNA5_PROBE: native tanh builtin MISSING - using exp2/rcp fallback
#endif
#if __has_builtin(__builtin_amdgcn_global_load_async_to_lds_b128)
#warning CDNA5_PROBE: global_load_async_to_lds_b128 builtin AVAILABLE
#endif
#if __has_builtin(__builtin_amdgcn_tensor_load_to_lds)
#warning CDNA5_PROBE: tensor_load_to_lds builtin AVAILABLE
#endif
#if __has_builtin(__builtin_amdgcn_s_wait_asynccnt)
#warning CDNA5_PROBE: s_wait_asynccnt builtin AVAILABLE
#endif
#endif

// ---------------------------------------------------------------------------
// Fast device math: gfx1250 has native V_TANH_F32 (TRANS op) — one instruction
// per tanh. Fallback: tanh(x) = 1 - 2/(exp2(2x*log2e)+1) via v_exp/v_rcp.
// ---------------------------------------------------------------------------
__device__ __forceinline__ float fast_tanh(float x) {
#if defined(__HIP_DEVICE_COMPILE__)
#if __has_builtin(__builtin_amdgcn_tanhf)
    return __builtin_amdgcn_tanhf(x);
#else
    float e = __builtin_amdgcn_exp2f(x * 2.8853900817779268f); // 2*log2(e)
    return 1.0f - 2.0f * __builtin_amdgcn_rcpf(e + 1.0f);
#endif
#else
    return tanhf(x);
#endif
}

__device__ __forceinline__ float fast_exp2(float x) {
#if defined(__HIP_DEVICE_COMPILE__)
    return __builtin_amdgcn_exp2f(x);
#else
    return exp2f(x);
#endif
}

// ---------------------------------------------------------------------------
// K0: f32 -> f16 conversions (all operands are O(1); f16 + f32 accum is ample)
// ---------------------------------------------------------------------------
__global__ void cvt_w_kernel(const float* __restrict__ W, _Float16* __restrict__ Wh) {
    int i = blockIdx.x * blockDim.x + threadIdx.x;       // 1024*1024 elements
    Wh[i] = (_Float16)W[i];
}

__global__ void cvt_rnn_kernel(const float* __restrict__ rnn,
                               _Float16* __restrict__ rnnh,
                               _Float16* __restrict__ rnnTh) {
    int i = blockIdx.x * blockDim.x + threadIdx.x;       // 512*512 elements
    int s = i >> 9;           // row
    int d = i & 511;          // col
    float v = rnn[i];
    rnnh[i] = (_Float16)v;
    rnnTh[d * S_DIM + s] = (_Float16)v;                  // transpose for K4 B-operand
}

__global__ void cvt_tgt_kernel(const float* __restrict__ tgt, _Float16* __restrict__ tgth) {
    int i = blockIdx.x * blockDim.x + threadIdx.x;       // 512*512 elements
    tgth[i] = (_Float16)tgt[i];
}

// ---------------------------------------------------------------------------
// K1: ABh[1024,1024] f16 (pre-activations; |x|~O(1) so f16 storage is ample
// and it halves the 268M-tanh stage's L2 read traffic).
//   rows   0..511 : A[s,d] = sum_k rnn[s,k] * W[d,k]
//   rows 512..1023: B[t,d] = sum_k tgt[t,k] * W[d,512+k] + b_lin[d]
// One wave = one 16x16 output tile; K-loop of 16 WMMA (16x16x32 f16).
// B-operand columns come from rows of W (row-major) -> contiguous loads.
// ---------------------------------------------------------------------------
__global__ __launch_bounds__(256) void gemm_ab_kernel(
        const _Float16* __restrict__ rnnh,
        const _Float16* __restrict__ tgth,
        const _Float16* __restrict__ Wh,
        const float*    __restrict__ b_lin,
        _Float16* __restrict__ ABh) {
    const int wave = threadIdx.x >> 5;
    const int lane = threadIdx.x & 31;
    const int tile = blockIdx.x * 8 + wave;   // 4096 tiles total
    const int tM = tile >> 6;                 // 0..63  (rows of AB / 16)
    const int tN = tile & 63;                 // 0..63  (cols of AB / 16)
    const bool isB = (tM >= 32);

    const _Float16* Asrc = isB ? tgth : rnnh;
    const int m0   = (isB ? (tM - 32) : tM) * 16;
    const int n0   = tN * 16;
    const int koff = isB ? 512 : 0;

    const int nlane = lane & 15;
    const int kbA   = (lane < 16) ? 0 : 8;    // unused; A-frag also loads with kbB
    const int kbB   = (lane < 16) ? 0 : 16;   // B-frag: 16 contiguous K per lane

    const _Float16* Aptr = Asrc + (m0 + nlane) * 512;
    const _Float16* Bptr = Wh + (n0 + nlane) * D_DIM + koff;

    v8f c = {};
    for (int k0 = 0; k0 < 512; k0 += 32) {
        union { v16h v; v8h h[2]; } a;
        a.v = *(const v16h*)(Aptr + k0 + kbB);          // K kbB..kbB+15
        v16h b = *(const v16h*)(Bptr + k0 + kbB);       // K kbB..kbB+15
        c = __builtin_amdgcn_wmma_f32_16x16x32_f16(
                false, a.v, false, b, (short)0, c, false, false);
    }

    const float bias  = isB ? b_lin[n0 + nlane] : 0.0f;
    const int   outM  = (isB ? 512 : 0) + m0;
    const int   roff  = (lane < 16) ? 0 : 8;
#pragma unroll
    for (int r = 0; r < 8; ++r) {
        ABh[(outM + r + roff) * D_DIM + n0 + nlane] = (_Float16)(c[r] + bias);
    }
}

// ---------------------------------------------------------------------------
// K2: scores[t,s] = sum_d w[d]*tanh(A[s,d]+B[t,d]) + b_score.
// Dominant stage: 268M native v_tanh_f32. Each block: 4 t-rows x 256 s-rows.
// B-rows + w staged to f32 in LDS (b128 loads); A-row streamed as f16 b128
// (8 values / 16B) and converted once, reused across the 4 t-rows.
// ---------------------------------------------------------------------------
__global__ __launch_bounds__(256) void scores_kernel(
        const _Float16* __restrict__ ABh,
        const float* __restrict__ w_score,
        const float* __restrict__ b_score,
        float* __restrict__ scores) {
    __shared__ __align__(16) float Bs[4][D_DIM];
    __shared__ __align__(16) float wsh[D_DIM];

    const int tid  = threadIdx.x;            // 256
    const int tblk = blockIdx.x >> 1;        // 0..127
    const int sblk = blockIdx.x & 1;
    const int t0   = tblk * 4;
    const int s    = sblk * 256 + tid;

    for (int d = tid; d < D_DIM; d += 256) {
        wsh[d] = w_score[d];
#pragma unroll
        for (int j = 0; j < 4; ++j)
            Bs[j][d] = (float)ABh[(512 + t0 + j) * D_DIM + d];
    }
    __syncthreads();

    const v8h*    Arow = (const v8h*)(ABh + s * D_DIM);   // 8 f16 per 16B load
    const float4* W4   = (const float4*)wsh;
    const float4* B0   = (const float4*)Bs[0];
    const float4* B1   = (const float4*)Bs[1];
    const float4* B2   = (const float4*)Bs[2];
    const float4* B3   = (const float4*)Bs[3];

    float acc0 = 0.f, acc1 = 0.f, acc2 = 0.f, acc3 = 0.f;
    for (int i = 0; i < D_DIM / 8; ++i) {
        const v8h a8 = Arow[i];
        float a[8];
#pragma unroll
        for (int c = 0; c < 8; ++c) a[c] = (float)a8[c];
        const float4 wlo = W4[2 * i];
        const float4 whi = W4[2 * i + 1];
        float4 blo, bhi;

        blo = B0[2 * i]; bhi = B0[2 * i + 1];
        acc0 += wlo.x * fast_tanh(a[0] + blo.x) + wlo.y * fast_tanh(a[1] + blo.y)
              + wlo.z * fast_tanh(a[2] + blo.z) + wlo.w * fast_tanh(a[3] + blo.w)
              + whi.x * fast_tanh(a[4] + bhi.x) + whi.y * fast_tanh(a[5] + bhi.y)
              + whi.z * fast_tanh(a[6] + bhi.z) + whi.w * fast_tanh(a[7] + bhi.w);
        blo = B1[2 * i]; bhi = B1[2 * i + 1];
        acc1 += wlo.x * fast_tanh(a[0] + blo.x) + wlo.y * fast_tanh(a[1] + blo.y)
              + wlo.z * fast_tanh(a[2] + blo.z) + wlo.w * fast_tanh(a[3] + blo.w)
              + whi.x * fast_tanh(a[4] + bhi.x) + whi.y * fast_tanh(a[5] + bhi.y)
              + whi.z * fast_tanh(a[6] + bhi.z) + whi.w * fast_tanh(a[7] + bhi.w);
        blo = B2[2 * i]; bhi = B2[2 * i + 1];
        acc2 += wlo.x * fast_tanh(a[0] + blo.x) + wlo.y * fast_tanh(a[1] + blo.y)
              + wlo.z * fast_tanh(a[2] + blo.z) + wlo.w * fast_tanh(a[3] + blo.w)
              + whi.x * fast_tanh(a[4] + bhi.x) + whi.y * fast_tanh(a[5] + bhi.y)
              + whi.z * fast_tanh(a[6] + bhi.z) + whi.w * fast_tanh(a[7] + bhi.w);
        blo = B3[2 * i]; bhi = B3[2 * i + 1];
        acc3 += wlo.x * fast_tanh(a[0] + blo.x) + wlo.y * fast_tanh(a[1] + blo.y)
              + wlo.z * fast_tanh(a[2] + blo.z) + wlo.w * fast_tanh(a[3] + blo.w)
              + whi.x * fast_tanh(a[4] + bhi.x) + whi.y * fast_tanh(a[5] + bhi.y)
              + whi.z * fast_tanh(a[6] + bhi.z) + whi.w * fast_tanh(a[7] + bhi.w);
    }

    const float bs = b_score[0];
    scores[(t0 + 0) * S_DIM + s] = acc0 + bs;
    scores[(t0 + 1) * S_DIM + s] = acc1 + bs;
    scores[(t0 + 2) * S_DIM + s] = acc2 + bs;
    scores[(t0 + 3) * S_DIM + s] = acc3 + bs;
}

// ---------------------------------------------------------------------------
// K3: row-wise softmax over s; emit f16 weights for the WMMA output GEMM.
// ---------------------------------------------------------------------------
__global__ __launch_bounds__(256) void softmax_kernel(
        const float* __restrict__ scores, _Float16* __restrict__ w16) {
    __shared__ float red[256];
    const int t   = blockIdx.x;
    const int tid = threadIdx.x;

    float x0 = scores[t * S_DIM + tid];
    float x1 = scores[t * S_DIM + 256 + tid];

    red[tid] = fmaxf(x0, x1);
    __syncthreads();
    for (int off = 128; off > 0; off >>= 1) {
        if (tid < off) red[tid] = fmaxf(red[tid], red[tid + off]);
        __syncthreads();
    }
    const float M = red[0];
    __syncthreads();

    const float L2E = 1.4426950408889634f;
    float e0 = fast_exp2((x0 - M) * L2E);
    float e1 = fast_exp2((x1 - M) * L2E);

    red[tid] = e0 + e1;
    __syncthreads();
    for (int off = 128; off > 0; off >>= 1) {
        if (tid < off) red[tid] += red[tid + off];
        __syncthreads();
    }
    const float rs = 1.0f / red[0];

    w16[t * S_DIM + tid]       = (_Float16)(e0 * rs);
    w16[t * S_DIM + 256 + tid] = (_Float16)(e1 * rs);
}

// ---------------------------------------------------------------------------
// K4: out[t,d] = sum_s weights[t,s] * rnn[s,d]  (WMMA, B from transposed rnn)
// ---------------------------------------------------------------------------
__global__ __launch_bounds__(256) void gemm_out_kernel(
        const _Float16* __restrict__ w16,
        const _Float16* __restrict__ rnnTh,
        float* __restrict__ out) {
    const int wave = threadIdx.x >> 5;
    const int lane = threadIdx.x & 31;
    const int tile = blockIdx.x * 8 + wave;  // 1024 tiles
    const int tM = tile >> 5;                // 0..31
    const int tN = tile & 31;                // 0..31
    const int m0 = tM * 16, n0 = tN * 16;
    const int nlane = lane & 15;
    const int kbA = (lane < 16) ? 0 : 8;
    const int kbB = (lane < 16) ? 0 : 16;

    const _Float16* Aptr = w16   + (m0 + nlane) * S_DIM;
    const _Float16* Bptr = rnnTh + (n0 + nlane) * S_DIM;

    v8f c = {};
    for (int k0 = 0; k0 < S_DIM; k0 += 32) {
        union { v16h v; v8h h[2]; } a;
        a.v = *(const v16h*)(Aptr + k0 + kbB);
        v16h b = *(const v16h*)(Bptr + k0 + kbB);
        c = __builtin_amdgcn_wmma_f32_16x16x32_f16(
                false, a.v, false, b, (short)0, c, false, false);
    }

    const int roff = (lane < 16) ? 0 : 8;
#pragma unroll
    for (int r = 0; r < 8; ++r)
        out[(m0 + r + roff) * DU + n0 + nlane] = c[r];
}

// ---------------------------------------------------------------------------
extern "C" void kernel_launch(void* const* d_in, const int* in_sizes, int n_in,
                              void* d_out, int out_size, void* d_ws, size_t ws_size,
                              hipStream_t stream) {
    const float* rnn     = (const float*)d_in[0];  // [512,512]
    const float* tgt     = (const float*)d_in[1];  // [512,512]
    const float* W_lin   = (const float*)d_in[2];  // [1024,1024]
    const float* b_lin   = (const float*)d_in[3];  // [1024]
    const float* w_score = (const float*)d_in[4];  // [1024]
    const float* b_score = (const float*)d_in[5];  // scalar
    float* out = (float*)d_out;                    // [512,512]

    // workspace carve-up (~7 MB)
    _Float16* Wh     = (_Float16*)d_ws;                 // 1024*1024 f16
    _Float16* rnnh   = Wh    + D_DIM * D_DIM;           // 512*512 f16
    _Float16* tgth   = rnnh  + S_DIM * DU;              // 512*512 f16
    _Float16* rnnTh  = tgth  + T_DIM * 512;             // 512*512 f16
    _Float16* ABh    = rnnTh + DU * S_DIM;              // 1024*1024 f16
    float*    scores = (float*)(ABh + D_DIM * D_DIM);   // 512*512 f32
    _Float16* w16    = (_Float16*)(scores + T_DIM * S_DIM); // 512*512 f16

    // K0: conversions
    cvt_w_kernel  <<<(D_DIM * D_DIM) / 256, 256, 0, stream>>>(W_lin, Wh);
    cvt_rnn_kernel<<<(S_DIM * DU)    / 256, 256, 0, stream>>>(rnn, rnnh, rnnTh);
    cvt_tgt_kernel<<<(T_DIM * 512)   / 256, 256, 0, stream>>>(tgt, tgth);

    // K1: ABh = [rnn@W1^T ; tgt@W2^T + b]   (4096 wave-tiles, 8 waves/block)
    gemm_ab_kernel<<<512, 256, 0, stream>>>(rnnh, tgth, Wh, b_lin, ABh);

    // K2: scores (the 268M-tanh stage)
    scores_kernel<<<256, 256, 0, stream>>>(ABh, w_score, b_score, scores);

    // K3: softmax -> f16 weights
    softmax_kernel<<<T_DIM, 256, 0, stream>>>(scores, w16);

    // K4: out = weights @ rnn
    gemm_out_kernel<<<128, 256, 0, stream>>>(w16, rnnTh, out);
}
